// MultiHeadGATLayer_53300544143790
// MI455X (gfx1250) — matrix operand order repaired and checked by reading, not compiled, not run on hardware
//
#include <hip/hip_runtime.h>
#include <hip/hip_bf16.h>

// MultiHeadGAT (N=4096, IN_F=512, OUT_F=64, HEADS=8) fused for gfx1250.
// Flash-style: attention matrix never materialized; adj read twice (128MB HBM),
// staged into LDS via the Tensor Data Mover (double-buffered, TENSORcnt).
// All matrix math on v_wmma_f32_16x16x32_f16, fp32 accumulate, wave32.

#define N_NODES 4096
#define IN_FEAT 512
#define OUT_FEAT 64
#define N_HEADS 8
#define LRELU_ALPHA 0.2f
#define MASK_NEG (-9.0e15f)
#define CT 128  // column tile for fused attention pass

typedef __attribute__((ext_vector_type(16))) _Float16 v16h;
typedef __attribute__((ext_vector_type(8)))  float    v8f;
typedef __attribute__((ext_vector_type(4))) unsigned int v4u;
typedef __attribute__((ext_vector_type(8)))  int      v8i;
typedef __attribute__((ext_vector_type(4)))  int      v4i;

union Frag16 {
    v16h v;
    _Float16 h[16];
    v4u q[2];
};

// ---- Tensor Data Mover: 2D tile (tile_d0 x tile_d1 of 4-byte elems) -> LDS ----
// D# built per CDNA5 ISA (group0/group1; groups 2-3 zero).
// This toolchain exposes the 6-arg builtin (extra int32x8 group before cpol).
__device__ __forceinline__ void tdm_load_2d_b32(unsigned lds_off, const void* gptr,
                                                unsigned tile_d0, unsigned tile_d1,
                                                unsigned tensor_d0, unsigned tensor_d1,
                                                unsigned stride0) {
    unsigned long long ga = (unsigned long long)(uintptr_t)gptr;
    v4u g0;
    g0.x = 1u;                                          // count=1, user load desc
    g0.y = lds_off;                                     // lds_addr (bytes)
    g0.z = (unsigned)(ga & 0xFFFFFFFFu);                // global_addr[31:0]
    g0.w = (unsigned)((ga >> 32) & 0x01FFFFFFu) | (2u << 30);  // addr[56:32], type=2
    v8i g1;
    g1[0] = (int)(2u << 16);                            // data_size=2 (4B), wg_mask=0
    g1[1] = (int)((tensor_d0 & 0xFFFFu) << 16);         // tensor_dim0[15:0]
    g1[2] = (int)((tensor_d0 >> 16) | ((tensor_d1 & 0xFFFFu) << 16));
    g1[3] = (int)((tensor_d1 >> 16) | ((tile_d0 & 0xFFFFu) << 16));
    g1[4] = (int)(tile_d1 & 0xFFFFu);                   // tile_dim1 (tile_dim2=0)
    g1[5] = (int)stride0;                               // tensor_dim0_stride[31:0]
    g1[6] = 0;                                          // stride0[47:32], stride1 lo
    g1[7] = 0;
    v4i g2 = {0, 0, 0, 0};
    v4i g3 = {0, 0, 0, 0};
    v8i g4 = {0, 0, 0, 0, 0, 0, 0, 0};
    __builtin_amdgcn_tensor_load_to_lds(g0, g1, g2, g3, g4, 0);
}

// ---------------------------------------------------------------- K0a: h -> f16
__global__ void convert_h_kernel(const float* __restrict__ h, _Float16* __restrict__ hb) {
    int idx = blockIdx.x * blockDim.x + threadIdx.x;  // N_NODES*IN_FEAT threads
    hb[idx] = (_Float16)h[idx];
}

// ------------------------------------------- K0b: W (h,k,o) -> f16 Wt (h,o,k)
__global__ void convert_w_kernel(const float* __restrict__ W, _Float16* __restrict__ Wt) {
    int idx = blockIdx.x * blockDim.x + threadIdx.x;  // N_HEADS*OUT_FEAT*IN_FEAT
    int k    = idx & (IN_FEAT - 1);
    int o    = (idx >> 9) & (OUT_FEAT - 1);
    int head = idx >> 15;
    Wt[idx] = (_Float16)W[(size_t)head * IN_FEAT * OUT_FEAT + (size_t)k * OUT_FEAT + o];
}

// ---------------------------------------------- K1: Wh = h @ W  (per head, WMMA)
// 8 waves/block; wave owns 16 rows x 64 cols. Emits Wh f32 and Wh^T f16.
__global__ void __launch_bounds__(256)
gemm_wh_kernel(const _Float16* __restrict__ hb, const _Float16* __restrict__ Wt,
               float* __restrict__ Whf, _Float16* __restrict__ WhT) {
    const int wave  = threadIdx.x >> 5;
    const int lane  = threadIdx.x & 31;
    const int head  = blockIdx.y;
    const int row0  = blockIdx.x * 128 + wave * 16;
    const int nlo   = lane & 15;
    const int khalf = lane >> 4;

    const _Float16* wslice = Wt + (size_t)head * OUT_FEAT * IN_FEAT;

    v8f acc[4] = {};
    Frag16 a, b;
    for (int kb = 0; kb < IN_FEAT; kb += 32) {
        // A fragment: 16x32 f16, lane = row (nlo), K split per ISA layout
        const _Float16* ap = hb + (size_t)(row0 + nlo) * IN_FEAT + kb + khalf * 8;
        a.q[0] = *(const v4u*)(ap);
        a.q[1] = *(const v4u*)(ap + 16);
#pragma unroll
        for (int t = 0; t < 4; ++t) {
            // B fragment: 32x16 f16, lane = col o, K split per ISA layout
            const _Float16* bp = wslice + (size_t)(t * 16 + nlo) * IN_FEAT + kb + khalf * 8;
            b.q[0] = *(const v4u*)(bp);
            b.q[1] = *(const v4u*)(bp + 16);
            acc[t] = __builtin_amdgcn_wmma_f32_16x16x32_f16(
                false, a.v, false, b.v, (short)0, acc[t], false, false);
        }
    }
    // C/D layout: lane -> col N = nlo, VGPR r -> row M = r + 8*khalf
    float*     whf = Whf + ((size_t)head * N_NODES + row0) * OUT_FEAT;
    _Float16*  wht = WhT + (size_t)head * OUT_FEAT * N_NODES;
#pragma unroll
    for (int t = 0; t < 4; ++t) {
        int o = t * 16 + nlo;
#pragma unroll
        for (int r = 0; r < 8; ++r) {
            int m = r + khalf * 8;
            float vv = acc[t][r];
            whf[(size_t)m * OUT_FEAT + o] = vv;
            wht[(size_t)o * N_NODES + row0 + m] = (_Float16)vv;
        }
    }
}

// ------------------------------------------------- K2: f1/f2 attention logits
__global__ void f12_kernel(const float* __restrict__ Whf, const float* __restrict__ a,
                           float* __restrict__ f1, float* __restrict__ f2) {
    int idx = blockIdx.x * blockDim.x + threadIdx.x;  // N_HEADS*N_NODES
    int head = idx >> 12;
    const float* row = Whf + (size_t)idx * OUT_FEAT;
    const float* a1  = a + (size_t)head * 2 * OUT_FEAT;
    const float* a2  = a1 + OUT_FEAT;
    float s1 = 0.f, s2 = 0.f;
#pragma unroll 8
    for (int o = 0; o < OUT_FEAT; ++o) {
        float w = row[o];
        s1 += w * a1[o];
        s2 += w * a2[o];
    }
    f1[idx] = s1;
    f2[idx] = s2;
}

// -------------------------- K3a: per-row online softmax stats, all 8 heads/block
// adj row (16KB) staged to LDS via TDM from wave 0; all 8 waves consume it.
__global__ void __launch_bounds__(256)
softmax_stats_kernel(const int* __restrict__ adj, const float* __restrict__ f1,
                     const float* __restrict__ f2, float* __restrict__ rowmax,
                     float* __restrict__ rowsum) {
    __shared__ int adj_row[N_NODES];
    const int row = blockIdx.x;
    const int tid = threadIdx.x;
    const int head = tid >> 5;
    const int lane = tid & 31;
    if (head == 0) {
        tdm_load_2d_b32((unsigned)(uintptr_t)&adj_row[0],
                        adj + (size_t)row * N_NODES,
                        /*tile*/ N_NODES, 1, /*tensor*/ N_NODES, N_NODES,
                        /*stride0*/ N_NODES);
        __builtin_amdgcn_s_wait_tensorcnt(0);
    }
    __syncthreads();
    const float f1v = f1[head * N_NODES + row];
    const float* f2h = f2 + head * N_NODES;
    float m = -3.0e38f, s = 0.0f;
    for (int j0 = 0; j0 < N_NODES; j0 += 32) {
        int j = j0 + lane;
        float e;
        if (adj_row[j] > 0) {
            float x = f1v + f2h[j];
            e = (x > 0.f) ? x : LRELU_ALPHA * x;
        } else {
            e = MASK_NEG;
        }
        float mn = fmaxf(m, e);
        s = s * __expf(m - mn) + __expf(e - mn);
        m = mn;
    }
#pragma unroll
    for (int off = 16; off > 0; off >>= 1) {
        float mo = __shfl_xor(m, off, 32);
        float so = __shfl_xor(s, off, 32);
        float mn = fmaxf(m, mo);
        s = s * __expf(m - mn) + so * __expf(mo - mn);
        m = mn;
    }
    if (lane == 0) {
        rowmax[head * N_NODES + row] = m;
        rowsum[head * N_NODES + row] = s;
    }
}

// ------------- K3b: fused softmax-apply + P@Wh (flash-style, WMMA), write output
// Block = 16 output rows x all 8 heads (wave w -> head w). adj tile (16xCT int32)
// is TDM-DMA'd into double-buffered LDS; f2 tile double-buffered via registers.
__global__ void __launch_bounds__(256)
attn_out_kernel(const int* __restrict__ adj, const float* __restrict__ f1,
                const float* __restrict__ f2, const float* __restrict__ rowmax,
                const float* __restrict__ rowsum, const _Float16* __restrict__ WhT,
                float* __restrict__ out) {
    __shared__ int   adj_t[2][16 * CT];    // 2 x 8 KB, TDM ping-pong
    __shared__ float f2_t[N_HEADS * CT];   // 4 KB
    const int tid   = threadIdx.x;
    const int head  = tid >> 5;
    const int lane  = tid & 31;
    const int nlo   = lane & 15;
    const int khalf = lane >> 4;
    const int row0  = blockIdx.x * 16;
    const int row   = row0 + nlo;

    const float f1r  = f1[head * N_NODES + row];
    const float rmax = rowmax[head * N_NODES + row];
    const float rinv = 1.0f / rowsum[head * N_NODES + row];
    const _Float16* whh = WhT + (size_t)head * OUT_FEAT * N_NODES;

    v8f acc[4] = {};
    Frag16 pa, pb;
    float f2reg[4];

    // ---- prologue: TDM adj tile 0, f2 tile 0 ----
    if (head == 0) {
        tdm_load_2d_b32((unsigned)(uintptr_t)&adj_t[0][0],
                        adj + (size_t)row0 * N_NODES,
                        CT, 16, N_NODES, N_NODES, N_NODES);
    }
#pragma unroll
    for (int i = 0; i < 4; ++i) {
        int idx = tid + 256 * i;                         // HEADS*CT = 1024
        f2reg[i] = f2[(idx / CT) * N_NODES + (idx % CT)];
    }
    if (head == 0) __builtin_amdgcn_s_wait_tensorcnt(0);
#pragma unroll
    for (int i = 0; i < 4; ++i) {
        int idx = tid + 256 * i;
        f2_t[idx] = f2reg[i];
    }
    __syncthreads();

    int cur = 0;
    for (int c0 = 0; c0 < N_NODES; c0 += CT) {
        const bool has_next = (c0 + CT) < N_NODES;
        if (has_next) {
            if (head == 0) {  // kick DMA of the next adj tile while we compute
                tdm_load_2d_b32((unsigned)(uintptr_t)&adj_t[cur ^ 1][0],
                                adj + (size_t)row0 * N_NODES + c0 + CT,
                                CT, 16, N_NODES, N_NODES, N_NODES);
            }
#pragma unroll
            for (int i = 0; i < 4; ++i) {  // next f2 tile -> registers (in flight)
                int idx = tid + 256 * i;
                f2reg[i] = f2[(idx / CT) * N_NODES + c0 + CT + (idx % CT)];
            }
        }
        const int* at = &adj_t[cur][0];
#pragma unroll
        for (int ks = 0; ks < CT; ks += 32) {
            // Build 16x32 probability A-fragment in registers.
            // Frag element f -> K = khalf*8 + (f&7) + ((f>>3)<<4)  (ISA layout)
#pragma unroll
            for (int f = 0; f < 16; ++f) {
                int K  = khalf * 8 + (f & 7) + ((f >> 3) << 4);
                int cc = ks + K;
                float e;
                if (at[nlo * CT + cc] > 0) {
                    float x = f1r + f2_t[head * CT + cc];
                    e = (x > 0.f) ? x : LRELU_ALPHA * x;
                } else {
                    e = MASK_NEG;  // exp underflows to exactly 0
                }
                pa.h[f] = (_Float16)(__expf(e - rmax) * rinv);
            }
#pragma unroll
            for (int t = 0; t < 4; ++t) {
                const _Float16* bp = whh + (size_t)(t * 16 + nlo) * N_NODES
                                         + c0 + ks + khalf * 8;
                pb.q[0] = *(const v4u*)(bp);
                pb.q[1] = *(const v4u*)(bp + 16);
                acc[t] = __builtin_amdgcn_wmma_f32_16x16x32_f16(
                    false, pa.v, false, pb.v, (short)0, acc[t], false, false);
            }
        }
        if (head == 0) __builtin_amdgcn_s_wait_tensorcnt(0);
        __syncthreads();                  // adj[cur^1] ready, f2_t fully consumed
        if (has_next) {
#pragma unroll
            for (int i = 0; i < 4; ++i) {
                int idx = tid + 256 * i;
                f2_t[idx] = f2reg[i];
            }
        }
        __syncthreads();                  // publish next f2 tile
        cur ^= 1;
    }

    // out[(row0+m)*512 + head*64 + o]  (reference: transpose(h',(1,0,2)).reshape)
#pragma unroll
    for (int t = 0; t < 4; ++t) {
        int o = t * 16 + nlo;
#pragma unroll
        for (int r = 0; r < 8; ++r) {
            int m = r + khalf * 8;
            out[(size_t)(row0 + m) * (N_HEADS * OUT_FEAT) + head * OUT_FEAT + o] = acc[t][r];
        }
    }
}

extern "C" void kernel_launch(void* const* d_in, const int* in_sizes, int n_in,
                              void* d_out, int out_size, void* d_ws, size_t ws_size,
                              hipStream_t stream) {
    (void)in_sizes; (void)n_in; (void)out_size; (void)ws_size;
    const float* h   = (const float*)d_in[0];
    const int*   adj = (const int*)d_in[1];
    const float* W   = (const float*)d_in[2];
    const float* a   = (const float*)d_in[3];
    float* out = (float*)d_out;

    char* ws = (char*)d_ws;
    size_t off = 0;
    auto carve = [&](size_t bytes) -> char* {
        char* p = ws + off;
        off += (bytes + 255) & ~(size_t)255;
        return p;
    };
    _Float16* hb   = (_Float16*)carve((size_t)N_NODES * IN_FEAT * 2);          // 4 MB
    _Float16* Wt   = (_Float16*)carve((size_t)N_HEADS * OUT_FEAT * IN_FEAT * 2);
    float*    Whf  = (float*)carve((size_t)N_HEADS * N_NODES * OUT_FEAT * 4);  // 8 MB
    _Float16* WhT  = (_Float16*)carve((size_t)N_HEADS * OUT_FEAT * N_NODES * 2);
    float*    f1   = (float*)carve((size_t)N_HEADS * N_NODES * 4);
    float*    f2   = (float*)carve((size_t)N_HEADS * N_NODES * 4);
    float*    rmax = (float*)carve((size_t)N_HEADS * N_NODES * 4);
    float*    rsum = (float*)carve((size_t)N_HEADS * N_NODES * 4);

    convert_h_kernel<<<(N_NODES * IN_FEAT) / 256, 256, 0, stream>>>(h, hb);
    convert_w_kernel<<<(N_HEADS * OUT_FEAT * IN_FEAT) / 256, 256, 0, stream>>>(W, Wt);
    gemm_wh_kernel<<<dim3(N_NODES / 128, N_HEADS), 256, 0, stream>>>(hb, Wt, Whf, WhT);
    f12_kernel<<<(N_HEADS * N_NODES) / 256, 256, 0, stream>>>(Whf, a, f1, f2);
    softmax_stats_kernel<<<N_NODES, 256, 0, stream>>>(adj, f1, f2, rmax, rsum);
    attn_out_kernel<<<N_NODES / 16, 256, 0, stream>>>(adj, f1, f2, rmax, rsum, WhT, out);
}
